// MultiScaleAttention_83416854823417
// MI455X (gfx1250) — matrix operand order repaired and checked by reading, not compiled, not run on hardware
//
#include <hip/hip_runtime.h>
#include <math.h>

typedef __bf16 bf16_t;
typedef __attribute__((ext_vector_type(8)))  __bf16 v8bf;
typedef __attribute__((ext_vector_type(16))) __bf16 v16bf;
typedef __attribute__((ext_vector_type(8)))  float  v8f;
typedef __attribute__((ext_vector_type(4)))  float  v4f;
typedef __attribute__((ext_vector_type(4)))  unsigned int u32x4;
typedef __attribute__((ext_vector_type(8)))  int    i32x8;
typedef __attribute__((ext_vector_type(4)))  int    i32x4;

#define BDIM   64
#define NTOK   256
#define HDIM   512
#define NHEADS 8
#define DKDIM  64
#define NBINS  3
#define MTOT   (BDIM * NTOK)          // 16384 token rows
#define LDSP   56                      // padded LDS row stride in bf16 (112B: 16B aligned, conflict-free)

#if defined(__has_builtin)
#  if __has_builtin(__builtin_amdgcn_tensor_load_to_lds) && __has_builtin(__builtin_amdgcn_s_wait_tensorcnt)
#    define HAVE_TDM 1
#  else
#    define HAVE_TDM 0
#  endif
#else
#  define HAVE_TDM 0
#endif

// ---------------------------------------------------------------------------
// fp32 -> bf16 elementwise convert
// ---------------------------------------------------------------------------
__global__ void k_cvt_bf16(const float* __restrict__ in, bf16_t* __restrict__ out, int n) {
    int i = blockIdx.x * 256 + threadIdx.x;
    if (i < n) out[i] = (bf16_t)in[i];
}

// ---------------------------------------------------------------------------
// WMMA fragment math shared by both cores: wave w covers 32x32 (2x2 frags)
// ---------------------------------------------------------------------------
__device__ __forceinline__ void wmma_step(const bf16_t* sA, const bf16_t* sB,
                                          int wm, int wn, int half, int mr,
                                          v8f c[2][2])
{
    v16bf a[2], bq[2];
    #pragma unroll
    for (int mt = 0; mt < 2; ++mt) {
        const int row = wm * 32 + mt * 16 + mr;
        v8bf lo = *(const v8bf*)&sA[row * LDSP + half * 8];
        v8bf hi = *(const v8bf*)&sA[row * LDSP + half * 8 + 16];
        a[mt] = __builtin_shufflevector(lo, hi, 0,1,2,3,4,5,6,7,8,9,10,11,12,13,14,15);
    }
    #pragma unroll
    for (int nt = 0; nt < 2; ++nt) {
        const int row = wn * 32 + nt * 16 + mr;
        v8bf lo = *(const v8bf*)&sB[row * LDSP + half * 8];
        v8bf hi = *(const v8bf*)&sB[row * LDSP + half * 8 + 16];
        bq[nt] = __builtin_shufflevector(lo, hi, 0,1,2,3,4,5,6,7,8,9,10,11,12,13,14,15);
    }
    #pragma unroll
    for (int mt = 0; mt < 2; ++mt)
        #pragma unroll
        for (int nt = 0; nt < 2; ++nt)
            c[mt][nt] = __builtin_amdgcn_wmma_f32_16x16x32_bf16(
                false, a[mt], false, bq[nt], (short)0, c[mt][nt], false, false);
}

#if HAVE_TDM
// ---------------------------------------------------------------------------
// Tensor Data Mover: async 2D tile (tile_w x tile_h bf16) -> LDS, row-padded
// 64B data + 48B pad per row => LDSP(56)-element LDS stride. ISA D# layout.
// ---------------------------------------------------------------------------
__device__ __forceinline__ void tdm_load_tile(unsigned lds_off, const bf16_t* g,
                                              unsigned lda, unsigned tile_w, unsigned tile_h)
{
    const unsigned long long ga = (unsigned long long)(uintptr_t)g;
    u32x4 g0;
    g0[0] = 1u;                                            // count=1 (valid load D#)
    g0[1] = lds_off;                                       // LDS byte address
    g0[2] = (unsigned)(ga & 0xffffffffu);                  // global addr [31:0]
    g0[3] = (unsigned)((ga >> 32) & 0x1ffffffu) | (2u << 30); // addr[56:32] | type=2
    const unsigned td = 0x40000000u;                       // huge tensor dims: tiles always in-bounds
    i32x8 g1;
    g1[0] = (int)((1u << 16)        // data_size = 2 bytes
                | (1u << 20)        // pad_enable
                | (3u << 22)        // pad_interval: 16 DWORDs (64B)
                | (11u << 25));     // pad_amount: 12 DWORDs (48B)
    g1[1] = (int)((td & 0xffffu) << 16);                   // tensor_dim0[15:0]
    g1[2] = (int)(((td >> 16) & 0xffffu) | ((td & 0xffffu) << 16)); // dim0 hi | dim1 lo
    g1[3] = (int)(((td >> 16) & 0xffffu) | (tile_w << 16));         // dim1 hi | tile_dim0
    g1[4] = (int)(tile_h & 0xffffu);                       // tile_dim1 | tile_dim2=0
    g1[5] = (int)lda;                                      // tensor_dim0_stride[31:0] (elements)
    g1[6] = 0;                                             // stride hi | dim1_stride lo
    g1[7] = 0;
    const i32x4 z4 = {0, 0, 0, 0};
    const i32x8 z8 = {0, 0, 0, 0, 0, 0, 0, 0};
    __builtin_amdgcn_tensor_load_to_lds(g0, g1, z4, z4, z8, 0);
}

// ---------------------------------------------------------------------------
// TDM-based double-buffered GEMM core: C[128x64] += A[128xK] * B'[64xK]^T
// blockDim.x == 256 (8 waves). Wave 0 drives the DMA; all waves do WMMA.
// ---------------------------------------------------------------------------
__device__ __forceinline__ void gemm_core_tdm(
    const bf16_t* __restrict__ Ab, int lda,
    const bf16_t* __restrict__ Bb, int ldb,
    int K, v8f c[2][2])
{
    __shared__ __align__(16) bf16_t sA[2][128 * LDSP];
    __shared__ __align__(16) bf16_t sB[2][64 * LDSP];

    const int tid  = threadIdx.x;
    const int lane = tid & 31;
    const int w    = tid >> 5;
    const int wm   = w & 3, wn = w >> 2;
    const int half = lane >> 4, mr = lane & 15;

    for (int i = 0; i < 2; ++i)
        for (int j = 0; j < 2; ++j)
            for (int e = 0; e < 8; ++e) c[i][j][e] = 0.0f;

    const unsigned ldsA[2] = { (unsigned)(uintptr_t)&sA[0][0], (unsigned)(uintptr_t)&sA[1][0] };
    const unsigned ldsB[2] = { (unsigned)(uintptr_t)&sB[0][0], (unsigned)(uintptr_t)&sB[1][0] };

    if (tid < 32) {                       // wave 0 issues DMA (TDM ignores EXEC)
        tdm_load_tile(ldsA[0], Ab, (unsigned)lda, 32u, 128u);
        tdm_load_tile(ldsB[0], Bb, (unsigned)ldb, 32u, 64u);
    }
    int nbuf = 0;
    for (int kk = 0; kk < K; kk += 32) {
        if (tid < 32) {
            if (kk + 32 < K) {            // prefetch next K-slice into other buffer
                tdm_load_tile(ldsA[nbuf ^ 1], Ab + kk + 32, (unsigned)lda, 32u, 128u);
                tdm_load_tile(ldsB[nbuf ^ 1], Bb + kk + 32, (unsigned)ldb, 32u, 64u);
                __builtin_amdgcn_s_wait_tensorcnt(2);  // in-order: current pair complete
            } else {
                __builtin_amdgcn_s_wait_tensorcnt(0);
            }
        }
        __syncthreads();                  // current buffer visible to all waves
        wmma_step(&sA[nbuf][0], &sB[nbuf][0], wm, wn, half, mr, c);
        __syncthreads();                  // buffer free for reuse (2 iters later)
        nbuf ^= 1;
    }
}
#endif  // HAVE_TDM

// ---------------------------------------------------------------------------
// Manual-staging GEMM core (fallback + fp32-A variant that converts to bf16)
// ---------------------------------------------------------------------------
template<bool AF32>
__device__ __forceinline__ void gemm_core_manual(
    const void* __restrict__ Ab, int lda,
    const bf16_t* __restrict__ Bb, int ldb,
    int K, v8f c[2][2])
{
    __shared__ __align__(16) bf16_t sA[128 * LDSP];
    __shared__ __align__(16) bf16_t sB[64 * LDSP];

    const int tid  = threadIdx.x;
    const int lane = tid & 31;
    const int w    = tid >> 5;
    const int wm   = w & 3, wn = w >> 2;
    const int half = lane >> 4, mr = lane & 15;

    for (int i = 0; i < 2; ++i)
        for (int j = 0; j < 2; ++j)
            for (int e = 0; e < 8; ++e) c[i][j][e] = 0.0f;

    for (int kk = 0; kk < K; kk += 32) {
        for (int ch = tid; ch < 512; ch += 256) {
            const int row = ch >> 2, c8 = (ch & 3) << 3;
            v8bf av;
            if constexpr (AF32) {
                const float* ap = (const float*)Ab + (size_t)row * lda + kk + c8;
                v4f f0 = *(const v4f*)ap;
                v4f f1 = *(const v4f*)(ap + 4);
                #pragma unroll
                for (int i = 0; i < 4; ++i) { av[i] = (bf16_t)f0[i]; av[4 + i] = (bf16_t)f1[i]; }
            } else {
                av = *(const v8bf*)((const bf16_t*)Ab + (size_t)row * lda + kk + c8);
            }
            *(v8bf*)(&sA[row * LDSP + c8]) = av;
        }
        {
            const int row = tid >> 2, c8 = (tid & 3) << 3;
            *(v8bf*)(&sB[row * LDSP + c8]) = *(const v8bf*)(Bb + (size_t)row * ldb + kk + c8);
            if (kk + 32 < K)
                __builtin_prefetch(Bb + (size_t)row * ldb + kk + 32 + c8, 0, 1);
        }
        __syncthreads();
        wmma_step(sA, sB, wm, wn, half, mr, c);
        __syncthreads();
    }
}

template<bool AF32>
__device__ __forceinline__ void gemm_core_128x64(
    const void* __restrict__ Ab, int lda,
    const bf16_t* __restrict__ Bb, int ldb,
    int K, v8f c[2][2])
{
#if HAVE_TDM
    if constexpr (!AF32) {
        gemm_core_tdm((const bf16_t*)Ab, lda, Bb, ldb, K, c);
    } else {
        gemm_core_manual<AF32>(Ab, lda, Bb, ldb, K, c);
    }
#else
    gemm_core_manual<AF32>(Ab, lda, Bb, ldb, K, c);
#endif
}

// C-fragment element coordinates (ISA: lanes 0-15 N, vgpr r: M=r / M=8+r)
#define C_LOOP_BEGIN \
    { const int lane_ = threadIdx.x & 31; const int w_ = threadIdx.x >> 5; \
      const int wm_ = w_ & 3, wn_ = w_ >> 2; \
      for (int mt = 0; mt < 2; ++mt) for (int nt = 0; nt < 2; ++nt) \
      for (int r = 0; r < 8; ++r) { \
        const int ml = wm_ * 32 + mt * 16 + ((lane_ >> 4) << 3) + r; \
        const int nl = wn_ * 32 + nt * 16 + (lane_ & 15); \
        float val = c[mt][nt][r];
#define C_LOOP_END } }

// ---------------------------------------------------------------------------
// QKV projection: x[16384,512] @ W^T + b -> head-split bf16 (mode2: V transposed)
// ---------------------------------------------------------------------------
__global__ void k_gemm_qkv(const bf16_t* __restrict__ xb, const bf16_t* __restrict__ wb,
                           const float* __restrict__ bias, bf16_t* __restrict__ dst,
                           int mode, float scale)
{
    const int m0 = blockIdx.x * 128, n0 = blockIdx.y * 64;
    v8f c[2][2];
    gemm_core_128x64<false>(xb + (size_t)m0 * HDIM, HDIM, wb + (size_t)n0 * HDIM, HDIM, HDIM, c);
    C_LOOP_BEGIN
        const int m = m0 + ml, n = n0 + nl;
        const float v = (val + bias[n]) * scale;
        const int b = m >> 8, t = m & 255, head = n >> 6, d = n & 63;
        if (mode == 2)  // V transposed: [b,h,d,t]
            dst[(((size_t)(b * NHEADS + head) * DKDIM) + d) * NTOK + t] = (bf16_t)v;
        else            // Q/K: [b,h,t,d]
            dst[(((size_t)(b * NHEADS + head) * NTOK) + t) * DKDIM + d] = (bf16_t)v;
    C_LOOP_END
}

// ---------------------------------------------------------------------------
// scores[bh,256,256] = q[bh] @ k[bh]^T  (q pre-scaled by 1/sqrt(dk))
// ---------------------------------------------------------------------------
__global__ void k_gemm_scores(const bf16_t* __restrict__ qh, const bf16_t* __restrict__ kh,
                              float* __restrict__ scores)
{
    const int tile = blockIdx.x;                    // 8 tiles: 2 M x 4 N
    const int m0 = (tile >> 2) * 128, n0 = (tile & 3) * 64;
    const int bh = blockIdx.z * NHEADS + blockIdx.y;
    const size_t qk_off = (size_t)bh * NTOK * DKDIM;
    v8f c[2][2];
    gemm_core_128x64<false>(qh + qk_off + (size_t)m0 * DKDIM, DKDIM,
                            kh + qk_off + (size_t)n0 * DKDIM, DKDIM, DKDIM, c);
    C_LOOP_BEGIN
        scores[(size_t)bh * NTOK * NTOK + (size_t)(m0 + ml) * NTOK + (n0 + nl)] = val;
    C_LOOP_END
}

// ---------------------------------------------------------------------------
// masked multi-bin softmax: one wave per (bin,b,h,row); writes attn into d_out
// ---------------------------------------------------------------------------
__global__ void k_softmax(const float* __restrict__ scores, const float* __restrict__ mask,
                          const float* __restrict__ dist,  const float* __restrict__ dist_bar,
                          float* __restrict__ attn_out)
{
    const int lane = threadIdx.x & 31;
    const int R = blockIdx.x * 8 + (threadIdx.x >> 5);
    const int n   = R & 255;
    const int h   = (R >> 8) & 7;
    const int b   = (R >> 11) & 63;
    const int bin = R >> 17;

    const float cut = dist_bar[b * NBINS + bin];
    const float* srow = scores + ((size_t)(b * NHEADS + h) * NTOK + n) * NTOK;
    const float* mrow = mask + (size_t)b * NTOK * NTOK + (size_t)n * NTOK;
    const float* drow = dist + (size_t)b * 255 * 255 + (size_t)(n - 1) * 255;

    float v[8];
    float mx = -INFINITY;
    #pragma unroll
    for (int j = 0; j < 8; ++j) {
        const int m = lane + j * 32;
        const float s = srow[m];
        bool dm = (n == 0) || (m == 0);
        if (!dm) dm = drow[m - 1] < cut;
        const float x = (!dm || mrow[m] == 0.0f) ? -1e12f : s;
        v[j] = x;
        mx = fmaxf(mx, x);
    }
    #pragma unroll
    for (int off = 16; off >= 1; off >>= 1) mx = fmaxf(mx, __shfl_xor(mx, off, 32));
    float sum = 0.0f;
    #pragma unroll
    for (int j = 0; j < 8; ++j) { v[j] = expf(v[j] - mx); sum += v[j]; }
    #pragma unroll
    for (int off = 16; off >= 1; off >>= 1) sum += __shfl_xor(sum, off, 32);
    const float inv = 1.0f / sum;

    float* arow = attn_out +
        ((((size_t)bin * BDIM + b) * NHEADS + h) * NTOK + n) * NTOK;
    #pragma unroll
    for (int j = 0; j < 8; ++j) arow[lane + j * 32] = v[j] * inv;
}

// ---------------------------------------------------------------------------
// ctx = attn(fp32, from d_out) @ v  -> bf16 [bin][B,N,H] (head-merged)
// ---------------------------------------------------------------------------
__global__ void k_gemm_ctx(const float* __restrict__ attn, const bf16_t* __restrict__ vt,
                           bf16_t* __restrict__ ctx)
{
    const int m0 = blockIdx.x * 128;
    const int bh = blockIdx.y;
    const int bin = blockIdx.z;
    v8f c[2][2];
    gemm_core_128x64<true>(
        attn + ((size_t)(bin * 512 + bh)) * NTOK * NTOK + (size_t)m0 * NTOK, NTOK,
        vt + (size_t)bh * DKDIM * NTOK, NTOK, NTOK, c);
    const int b = bh >> 3, h = bh & 7;
    C_LOOP_BEGIN
        if (nl < DKDIM) {
            const int t = m0 + ml;
            ctx[(size_t)bin * MTOT * HDIM + ((size_t)(b * NTOK + t)) * HDIM + h * DKDIM + nl]
                = (bf16_t)val;
        }
    C_LOOP_END
}

// ---------------------------------------------------------------------------
// generic bf16-out GEMM: out[m, colofs+n] = act(A @ B'^T + bias)
// ---------------------------------------------------------------------------
template<bool SILU>
__global__ void k_gemm_bf16out(const bf16_t* __restrict__ A, int lda,
                               const bf16_t* __restrict__ Bw, int ldb,
                               const float* __restrict__ bias,
                               bf16_t* __restrict__ out, int ldo, int colofs, int K)
{
    const int m0 = blockIdx.x * 128, n0 = blockIdx.y * 64;
    v8f c[2][2];
    gemm_core_128x64<false>(A + (size_t)m0 * lda, lda, Bw + (size_t)n0 * ldb, ldb, K, c);
    C_LOOP_BEGIN
        float v = val + bias[n0 + nl];
        if constexpr (SILU) v = v / (1.0f + expf(-v));
        out[(size_t)(m0 + ml) * ldo + colofs + n0 + nl] = (bf16_t)v;
    C_LOOP_END
}

// ---------------------------------------------------------------------------
// final GEMM (Ws2) + bias + residual -> fp32 h
// ---------------------------------------------------------------------------
__global__ void k_gemm_out(const bf16_t* __restrict__ A, const bf16_t* __restrict__ Bw,
                           const float* __restrict__ bias, const float* __restrict__ resid,
                           float* __restrict__ out)
{
    const int m0 = blockIdx.x * 128, n0 = blockIdx.y * 64;
    v8f c[2][2];
    gemm_core_128x64<false>(A + (size_t)m0 * HDIM, HDIM, Bw + (size_t)n0 * HDIM, HDIM, HDIM, c);
    C_LOOP_BEGIN
        const size_t idx = (size_t)(m0 + ml) * HDIM + n0 + nl;
        out[idx] = val + bias[n0 + nl] + resid[idx];
    C_LOOP_END
}

// ---------------------------------------------------------------------------
// LayerNorm over H=512: one wave per row (16 elems/lane), shuffle reductions
// ---------------------------------------------------------------------------
__global__ void k_layernorm(const float* __restrict__ h, const float* __restrict__ gamma,
                            const float* __restrict__ beta, float* __restrict__ out)
{
    const int lane = threadIdx.x & 31;
    const int row = blockIdx.x * 8 + (threadIdx.x >> 5);
    const float* hr = h + (size_t)row * HDIM;
    float x[16], s = 0.0f, ss = 0.0f;
    #pragma unroll
    for (int i = 0; i < 16; ++i) {
        x[i] = hr[lane + i * 32];
        s += x[i]; ss += x[i] * x[i];
    }
    #pragma unroll
    for (int off = 16; off >= 1; off >>= 1) { s += __shfl_xor(s, off, 32); ss += __shfl_xor(ss, off, 32); }
    const float mean = s * (1.0f / HDIM);
    const float var  = ss * (1.0f / HDIM) - mean * mean;
    const float inv  = rsqrtf(var + 1e-6f);
    float* orow = out + (size_t)row * HDIM;
    #pragma unroll
    for (int i = 0; i < 16; ++i) {
        const int col = lane + i * 32;
        orow[col] = (x[i] - mean) * inv * gamma[col] + beta[col];
    }
}

// ---------------------------------------------------------------------------
extern "C" void kernel_launch(void* const* d_in, const int* in_sizes, int n_in,
                              void* d_out, int out_size, void* d_ws, size_t ws_size,
                              hipStream_t stream)
{
    (void)in_sizes; (void)n_in; (void)out_size; (void)ws_size;
    const float* x        = (const float*)d_in[0];
    const float* dist     = (const float*)d_in[1];
    const float* dist_bar = (const float*)d_in[2];
    const float* mask     = (const float*)d_in[3];
    const float* Wq = (const float*)d_in[4];  const float* bq = (const float*)d_in[5];
    const float* Wk = (const float*)d_in[6];  const float* bk = (const float*)d_in[7];
    const float* Wv = (const float*)d_in[8];  const float* bv = (const float*)d_in[9];
    const float* Wa = (const float*)d_in[10]; const float* ba = (const float*)d_in[11];
    const float* Ws1 = (const float*)d_in[12]; const float* bs1 = (const float*)d_in[13];
    const float* Ws2 = (const float*)d_in[14]; const float* bs2 = (const float*)d_in[15];
    const float* gamma = (const float*)d_in[16]; const float* beta = (const float*)d_in[17];

    // workspace carve-up
    char* ws = (char*)d_ws;
    size_t cur = 0;
    auto alloc = [&](size_t bytes) { size_t r = cur; cur += (bytes + 255) & ~(size_t)255; return r; };
    const size_t XN = (size_t)MTOT * HDIM;            // 8,388,608
    const size_t W2 = (size_t)HDIM * HDIM;            // 262,144
    const size_t QH = (size_t)BDIM * NHEADS * NTOK * DKDIM; // 8,388,608
    const size_t SC = (size_t)BDIM * NHEADS * NTOK * NTOK;  // 33,554,432

    bf16_t* xb   = (bf16_t*)(ws + alloc(XN * 2));
    bf16_t* wqb  = (bf16_t*)(ws + alloc(W2 * 2));
    bf16_t* wkb  = (bf16_t*)(ws + alloc(W2 * 2));
    bf16_t* wvb  = (bf16_t*)(ws + alloc(W2 * 2));
    bf16_t* wab  = (bf16_t*)(ws + alloc(NBINS * W2 * 2));
    bf16_t* ws1b = (bf16_t*)(ws + alloc((size_t)HDIM * (NBINS * HDIM) * 2));
    bf16_t* ws2b = (bf16_t*)(ws + alloc(W2 * 2));
    bf16_t* qh   = (bf16_t*)(ws + alloc(QH * 2));
    bf16_t* kh   = (bf16_t*)(ws + alloc(QH * 2));
    bf16_t* vt   = (bf16_t*)(ws + alloc(QH * 2));
    float*  sc   = (float*) (ws + alloc(SC * 4));
    bf16_t* ctx  = (bf16_t*)(ws + alloc((size_t)NBINS * XN * 2));
    bf16_t* xcat = (bf16_t*)(ws + alloc((size_t)MTOT * NBINS * HDIM * 2));
    bf16_t* h1   = (bf16_t*)(ws + alloc(XN * 2));
    float*  hbuf = (float*) (ws + alloc(XN * 4));

    float* out_f    = (float*)d_out;
    float* attn_out = out_f + XN;

    auto cvt = [&](const float* src, bf16_t* dst, size_t n) {
        k_cvt_bf16<<<(unsigned)((n + 255) / 256), 256, 0, stream>>>(src, dst, (int)n);
    };
    cvt(x, xb, XN);
    cvt(Wq, wqb, W2); cvt(Wk, wkb, W2); cvt(Wv, wvb, W2);
    cvt(Wa, wab, NBINS * W2);
    cvt(Ws1, ws1b, (size_t)HDIM * NBINS * HDIM);
    cvt(Ws2, ws2b, W2);

    const dim3 blk(256);
    const float qscale = 0.125f;  // 1/sqrt(dk=64)

    k_gemm_qkv<<<dim3(128, 8), blk, 0, stream>>>(xb, wqb, bq, qh, 0, qscale);
    k_gemm_qkv<<<dim3(128, 8), blk, 0, stream>>>(xb, wkb, bk, kh, 1, 1.0f);
    k_gemm_qkv<<<dim3(128, 8), blk, 0, stream>>>(xb, wvb, bv, vt, 2, 1.0f);

    k_gemm_scores<<<dim3(8, NHEADS, BDIM), blk, 0, stream>>>(qh, kh, sc);

    k_softmax<<<(NBINS * BDIM * NHEADS * NTOK) / 8, blk, 0, stream>>>(
        sc, mask, dist, dist_bar, attn_out);

    k_gemm_ctx<<<dim3(2, BDIM * NHEADS, NBINS), blk, 0, stream>>>(attn_out, vt, ctx);

    for (int bin = 0; bin < NBINS; ++bin)
        k_gemm_bf16out<false><<<dim3(128, 8), blk, 0, stream>>>(
            ctx + (size_t)bin * XN, HDIM, wab + (size_t)bin * W2, HDIM,
            ba + bin * HDIM, xcat, NBINS * HDIM, bin * HDIM, HDIM);

    k_gemm_bf16out<true><<<dim3(128, 8), blk, 0, stream>>>(
        xcat, NBINS * HDIM, ws1b, NBINS * HDIM, bs1, h1, HDIM, 0, NBINS * HDIM);

    k_gemm_out<<<dim3(128, 8), blk, 0, stream>>>(h1, ws2b, bs2, x, hbuf);

    k_layernorm<<<MTOT / 8, blk, 0, stream>>>(hbuf, gamma, beta, out_f);
}
